// CausalSelfAttention_6897717477391
// MI455X (gfx1250) — compile-verified, matrix operand order checked by reading, not run on hardware
//
#include <hip/hip_runtime.h>
#include <stdint.h>

// ---------------- problem constants ----------------
#define BB    2
#define SS    2048
#define EE    1024
#define HH    16
#define DD    64
#define NQKV  (HH * 3 * DD)   // 3072
#define MROWS (BB * SS)       // 4096

typedef __attribute__((ext_vector_type(16))) _Float16     v16h;
typedef __attribute__((ext_vector_type(8)))  _Float16     v8h;
typedef __attribute__((ext_vector_type(8)))  float        v8f;
typedef __attribute__((ext_vector_type(4)))  unsigned int v4u;
typedef __attribute__((ext_vector_type(8)))  int          v8i;
typedef __attribute__((ext_vector_type(4)))  int          v4i;

// ---------------- TDM availability ----------------
#if defined(__has_builtin)
#if __has_builtin(__builtin_amdgcn_tensor_load_to_lds) && \
    __has_builtin(__builtin_amdgcn_s_wait_tensorcnt)
#define USE_TDM 1
#endif
#endif
#ifndef USE_TDM
#define USE_TDM 0
#endif

// ---------------- WMMA fragment helpers (CDNA5 wave32 layouts) ----------------
// A (16xK) / B (Kx16) f16 operand, lane L: lane-dim index = L&15,
// element i -> k = 8*(L>=16) + 16*(i>=8) + (i&7)  (two contiguous 8-half runs)

__device__ __forceinline__ v16h load_frag_contig(const _Float16* __restrict__ base,
                                                 int ld, int lane) {
  const int li  = lane & 15;
  const int hi8 = (lane >> 4) << 3;
  const _Float16* p = base + (size_t)li * ld + hi8;
  v8h lo = *(const v8h*)(p);
  v8h hi = *(const v8h*)(p + 16);
  v16h r;
#pragma unroll
  for (int i = 0; i < 8; ++i) { r[i] = lo[i]; r[i + 8] = hi[i]; }
  return r;
}

// A fragment gathered straight from an f32 row-major matrix (k contiguous)
__device__ __forceinline__ v16h gather_a_f32(const float* __restrict__ base,
                                             int ld, int lane) {
  const int li  = lane & 15;
  const int hi8 = (lane >> 4) << 3;
  const float* p = base + (size_t)li * ld + hi8;
  const float4 f0 = *(const float4*)(p);
  const float4 f1 = *(const float4*)(p + 4);
  const float4 f2 = *(const float4*)(p + 16);
  const float4 f3 = *(const float4*)(p + 20);
  v16h r;
  r[0]  = (_Float16)f0.x; r[1]  = (_Float16)f0.y; r[2]  = (_Float16)f0.z; r[3]  = (_Float16)f0.w;
  r[4]  = (_Float16)f1.x; r[5]  = (_Float16)f1.y; r[6]  = (_Float16)f1.z; r[7]  = (_Float16)f1.w;
  r[8]  = (_Float16)f2.x; r[9]  = (_Float16)f2.y; r[10] = (_Float16)f2.z; r[11] = (_Float16)f2.w;
  r[12] = (_Float16)f3.x; r[13] = (_Float16)f3.y; r[14] = (_Float16)f3.z; r[15] = (_Float16)f3.w;
  return r;
}

// B fragment gathered from an f32 row-major matrix (k strided by ld).
// For a fixed element i the 16 lanes of a half read 16 consecutive floats.
__device__ __forceinline__ v16h gather_b_f32(const float* __restrict__ base,
                                             int ld, int lane) {
  const int li  = lane & 15;
  const int hi8 = (lane >> 4) << 3;
  v16h r;
#pragma unroll
  for (int i = 0; i < 16; ++i) {
    const int k = hi8 + ((i >> 3) << 4) + (i & 7);
    r[i] = (_Float16)base[(size_t)k * ld + li];
  }
  return r;
}

__device__ __forceinline__ v8f wmma16x16x32(v16h a, v16h b, v8f c) {
  return __builtin_amdgcn_wmma_f32_16x16x32_f16(false, a, false, b, (short)0, c,
                                                false, false);
}

#if USE_TDM
// 2-D TDM load: tile (tile_d1 rows x tile_d0 elems) of 2-byte data into LDS.
// Descriptor bit layout per CDNA5 ISA 8.3/8.4 (group0 128b, group1 256b).
// 6-arg builtin: (uint32x4 g0, int32x8 g1, int32x4, int32x4, int32x8, i32 cpol)
__device__ __forceinline__ void tdm_load_2d(uint32_t lds_off, const void* gaddr,
                                            uint32_t tensor_d0, uint32_t tensor_d1,
                                            uint32_t stride0, uint32_t tile_d0,
                                            uint32_t tile_d1) {
  const uint64_t ga = (uint64_t)(uintptr_t)gaddr;
  v4u g0;
  g0[0] = 1u;                                               // count=1, user desc
  g0[1] = lds_off;                                          // lds_addr [63:32]
  g0[2] = (uint32_t)(ga & 0xffffffffu);                     // global_addr lo
  g0[3] = (uint32_t)((ga >> 32) & 0x01ffffffu) | (2u << 30);// ga hi + type=2
  v8i g1;
  g1[0] = (int)(1u << 16);                                  // data_size=1 (2B)
  g1[1] = (int)(tensor_d0 << 16);                           // tensor_dim0 lo16
  g1[2] = (int)((tensor_d0 >> 16) | (tensor_d1 << 16));     // td0 hi / td1 lo
  g1[3] = (int)((tensor_d1 >> 16) | (tile_d0 << 16));       // td1 hi / tile0
  g1[4] = (int)(tile_d1 & 0xffffu);                         // tile1, tile2=0
  g1[5] = (int)stride0;                                     // dim0 stride lo32
  g1[6] = 0;                                                // stride hi, d1str
  g1[7] = 0;
  const v4i z4 = {0, 0, 0, 0};
  const v8i z8 = {0, 0, 0, 0, 0, 0, 0, 0};
  __builtin_amdgcn_tensor_load_to_lds(g0, g1, z4, z4, z8, 0);
}
#endif

// =====================================================================
// Kernel 1: QKV projection GEMM  [4096 x 1024] * [1024 x 3072]
// 192 threads = 6 waves; block covers one head's 192 columns x 32 rows,
// each wave a 32x32 tile (2x2 WMMA, fragment reuse), no LDS, no division.
// Epilogue: Q (x 1/8) and K as [B*H][S][D] f16; V transposed [B*H][D][S].
// =====================================================================
__global__ __launch_bounds__(192) void qkv_gemm_kernel(
    const float* __restrict__ x, const float* __restrict__ w,
    _Float16* __restrict__ Qo, _Float16* __restrict__ Ko,
    _Float16* __restrict__ Vto) {
  const int tid  = threadIdx.x;
  const int lane = tid & 31;
  const int wv   = tid >> 5;              // 0..5
  const int m0   = blockIdx.y * 32;
  const int h    = blockIdx.x;            // head
  const int n0   = h * 192 + wv * 32;     // global output column base

  v8f a00 = {0,0,0,0,0,0,0,0}, a01 = a00, a10 = a00, a11 = a00;

  for (int kk = 0; kk < EE; kk += 32) {
    if (kk + 32 < EE) {
      __builtin_prefetch(w + (size_t)(kk + 32) * NQKV + n0 + lane, 0, 1);
    }
    const v16h fa0 = gather_a_f32(x + (size_t)m0 * EE + kk,        EE, lane);
    const v16h fa1 = gather_a_f32(x + (size_t)(m0 + 16) * EE + kk, EE, lane);
    const v16h fb0 = gather_b_f32(w + (size_t)kk * NQKV + n0,      NQKV, lane);
    const v16h fb1 = gather_b_f32(w + (size_t)kk * NQKV + n0 + 16, NQKV, lane);
    a00 = wmma16x16x32(fa0, fb0, a00);
    a01 = wmma16x16x32(fa0, fb1, a01);
    a10 = wmma16x16x32(fa1, fb0, a10);
    a11 = wmma16x16x32(fa1, fb1, a11);
  }

  const int col = lane & 15, hi = lane >> 4;
  const v8f accs[2][2] = {{a00, a01}, {a10, a11}};
#pragma unroll
  for (int im = 0; im < 2; ++im) {
#pragma unroll
    for (int in = 0; in < 2; ++in) {
      const int cl  = wv * 32 + in * 16;  // 0..176, 16-aligned -> one sel each
      const int sel = cl >> 6;            // 0=Q, 1=K, 2=V
      const int d   = (cl & 63) + col;    // head-dim index
#pragma unroll
      for (int r = 0; r < 8; ++r) {
        const int row = m0 + im * 16 + r + hi * 8;
        const int b_  = row >> 11;
        const int s   = row & 2047;
        const size_t bh = (size_t)(b_ * HH + h);
        const float v = accs[im][in][r];
        if (sel == 0)      Qo[(bh * SS + s) * DD + d]  = (_Float16)(v * 0.125f);
        else if (sel == 1) Ko[(bh * SS + s) * DD + d]  = (_Float16)v;
        else               Vto[(bh * DD + d) * SS + s] = (_Float16)v;
      }
    }
  }
}

// =====================================================================
// Kernel 2: causal flash attention. 1 wave per 16-query tile.
// grid = (S/16, B*H). K/V tiles double-buffered in LDS via the Tensor
// Data Mover: next block's DMA overlaps current block's WMMA work;
// s_wait_tensorcnt(2) retires just the current buffer (TDM is in-order).
// =====================================================================
__global__ __launch_bounds__(32) void flash_attn_kernel(
    const _Float16* __restrict__ Q, const _Float16* __restrict__ K,
    const _Float16* __restrict__ Vt, _Float16* __restrict__ O) {
#if USE_TDM
  __shared__ alignas(16) _Float16 Klds[2][32 * DD];  // [32 keys][64 d]
  __shared__ alignas(16) _Float16 Vlds[2][DD * 32];  // [64 d][32 keys]
#endif
  __shared__ alignas(16) _Float16 Plds[16 * 32];

  const int lane = threadIdx.x;
  const int qb   = blockIdx.x;
  const int bh   = blockIdx.y;
  const int q0   = qb * 16;
  const int col  = lane & 15;
  const int hi   = lane >> 4;

  const _Float16* Qb = Q  + (size_t)bh * SS * DD;
  const _Float16* Kb = K  + (size_t)bh * SS * DD;
  const _Float16* Vb = Vt + (size_t)bh * DD * SS;

  const v16h qa0 = load_frag_contig(Qb + (size_t)q0 * DD,      DD, lane);
  const v16h qa1 = load_frag_contig(Qb + (size_t)q0 * DD + 32, DD, lane);

  v8f o0 = {0,0,0,0,0,0,0,0}, o1 = o0, o2 = o0, o3 = o0;
  float mrow[8], lrow[8];
#pragma unroll
  for (int r = 0; r < 8; ++r) { mrow[r] = -1.0e30f; lrow[r] = 0.f; }

  const int nkb = (q0 >> 5) + 1;

#if USE_TDM
  auto issue_kv = [&](int kblk, int buf) {
    const int kk0 = kblk * 32;
    tdm_load_2d((uint32_t)(uintptr_t)&Klds[buf][0], Kb + (size_t)kk0 * DD,
                /*td0=*/DD, /*td1=*/SS, /*stride0=*/DD, /*tile0=*/DD, /*tile1=*/32);
    tdm_load_2d((uint32_t)(uintptr_t)&Vlds[buf][0], Vb + kk0,
                /*td0=*/SS, /*td1=*/DD, /*stride0=*/SS, /*tile0=*/32, /*tile1=*/DD);
  };
  issue_kv(0, 0);  // prologue
#endif

  for (int kb = 0; kb < nkb; ++kb) {
    const int k0 = kb * 32;

#if USE_TDM
    if (kb + 1 < nkb) {
      issue_kv(kb + 1, (kb + 1) & 1);      // overlap next DMA with compute
      __builtin_amdgcn_s_wait_tensorcnt(2);// current buffer's 2 ops retired
    } else {
      __builtin_amdgcn_s_wait_tensorcnt(0);
    }
    const _Float16* Kt = &Klds[kb & 1][0];
    const _Float16* Vtile = &Vlds[kb & 1][0];
#endif

    v8f s0 = {0,0,0,0,0,0,0,0}, s1 = s0;
#if USE_TDM
    s0 = wmma16x16x32(qa0, load_frag_contig(Kt,                DD, lane), s0);
    s0 = wmma16x16x32(qa1, load_frag_contig(Kt + 32,           DD, lane), s0);
    s1 = wmma16x16x32(qa0, load_frag_contig(Kt + 16 * DD,      DD, lane), s1);
    s1 = wmma16x16x32(qa1, load_frag_contig(Kt + 16 * DD + 32, DD, lane), s1);
#else
    if (kb + 1 < nkb) {
      __builtin_prefetch(Kb + (size_t)(k0 + 32) * DD + lane * 4, 0, 1);
      __builtin_prefetch(Vb + (size_t)(lane * 2) * SS + k0 + 32, 0, 1);
    }
    s0 = wmma16x16x32(qa0, load_frag_contig(Kb + (size_t)k0 * DD,             DD, lane), s0);
    s0 = wmma16x16x32(qa1, load_frag_contig(Kb + (size_t)k0 * DD + 32,        DD, lane), s0);
    s1 = wmma16x16x32(qa0, load_frag_contig(Kb + (size_t)(k0 + 16) * DD,      DD, lane), s1);
    s1 = wmma16x16x32(qa1, load_frag_contig(Kb + (size_t)(k0 + 16) * DD + 32, DD, lane), s1);
#endif

    // online softmax per row (C row = r + 8*hi, col = lane&15)
#pragma unroll
    for (int r = 0; r < 8; ++r) {
      const int qrow = q0 + r + hi * 8;
      float v0 = (k0 + col      <= qrow) ? s0[r] : -1.0e30f;
      float v1 = (k0 + 16 + col <= qrow) ? s1[r] : -1.0e30f;
      float mx = fmaxf(v0, v1);
#pragma unroll
      for (int off = 1; off < 16; off <<= 1)
        mx = fmaxf(mx, __shfl_xor(mx, off, 32));
      const float mnew  = fmaxf(mrow[r], mx);
      const float scale = __expf(mrow[r] - mnew);
      mrow[r] = mnew;
      const float p0 = __expf(v0 - mnew);
      const float p1 = __expf(v1 - mnew);
      float ps = p0 + p1;
#pragma unroll
      for (int off = 1; off < 16; off <<= 1)
        ps += __shfl_xor(ps, off, 32);
      lrow[r] = lrow[r] * scale + ps;
      o0[r] *= scale; o1[r] *= scale; o2[r] *= scale; o3[r] *= scale;
      Plds[(r + hi * 8) * 32 + col]      = (_Float16)p0;
      Plds[(r + hi * 8) * 32 + 16 + col] = (_Float16)p1;
    }
    __syncthreads();
    const v16h pa = load_frag_contig(Plds, 32, lane);
#if USE_TDM
    o0 = wmma16x16x32(pa, load_frag_contig(Vtile +  0 * 32, 32, lane), o0);
    o1 = wmma16x16x32(pa, load_frag_contig(Vtile + 16 * 32, 32, lane), o1);
    o2 = wmma16x16x32(pa, load_frag_contig(Vtile + 32 * 32, 32, lane), o2);
    o3 = wmma16x16x32(pa, load_frag_contig(Vtile + 48 * 32, 32, lane), o3);
#else
    o0 = wmma16x16x32(pa, load_frag_contig(Vb + (size_t)( 0) * SS + k0, SS, lane), o0);
    o1 = wmma16x16x32(pa, load_frag_contig(Vb + (size_t)(16) * SS + k0, SS, lane), o1);
    o2 = wmma16x16x32(pa, load_frag_contig(Vb + (size_t)(32) * SS + k0, SS, lane), o2);
    o3 = wmma16x16x32(pa, load_frag_contig(Vb + (size_t)(48) * SS + k0, SS, lane), o3);
#endif
    __syncthreads();
  }

  // normalize and store attnOut [B][S][H][D] f16
  const int b_ = bh >> 4, h = bh & 15;
#pragma unroll
  for (int r = 0; r < 8; ++r) {
    const int m = r + hi * 8;
    const float inv = 1.0f / lrow[r];
    const size_t base = (((size_t)b_ * SS + q0 + m) * HH + h) * DD + col;
    O[base +  0] = (_Float16)(o0[r] * inv);
    O[base + 16] = (_Float16)(o1[r] * inv);
    O[base + 32] = (_Float16)(o2[r] * inv);
    O[base + 48] = (_Float16)(o3[r] * inv);
  }
}

// =====================================================================
// Kernel 3: output projection GEMM  [4096 x 1024](f16) * [1024 x 1024](f32)
// Register-tiled WMMA scheme; f32 result straight to d_out.
// =====================================================================
__global__ __launch_bounds__(256) void out_gemm_kernel(
    const _Float16* __restrict__ A, const float* __restrict__ w,
    float* __restrict__ out) {
  const int tid  = threadIdx.x;
  const int lane = tid & 31;
  const int wv   = tid >> 5;
  const int m0   = blockIdx.y * 32;
  const int n0   = blockIdx.x * 256 + wv * 32;

  v8f a00 = {0,0,0,0,0,0,0,0}, a01 = a00, a10 = a00, a11 = a00;

  for (int kk = 0; kk < EE; kk += 32) {
    if (kk + 32 < EE) {
      __builtin_prefetch(w + (size_t)(kk + 32) * EE + n0 + lane, 0, 1);
    }
    const v16h fa0 = load_frag_contig(A + (size_t)m0 * EE + kk,        EE, lane);
    const v16h fa1 = load_frag_contig(A + (size_t)(m0 + 16) * EE + kk, EE, lane);
    const v16h fb0 = gather_b_f32(w + (size_t)kk * EE + n0,      EE, lane);
    const v16h fb1 = gather_b_f32(w + (size_t)kk * EE + n0 + 16, EE, lane);
    a00 = wmma16x16x32(fa0, fb0, a00);
    a01 = wmma16x16x32(fa0, fb1, a01);
    a10 = wmma16x16x32(fa1, fb0, a10);
    a11 = wmma16x16x32(fa1, fb1, a11);
  }

  const int col = lane & 15, hi = lane >> 4;
  const v8f accs[2][2] = {{a00, a01}, {a10, a11}};
#pragma unroll
  for (int im = 0; im < 2; ++im) {
#pragma unroll
    for (int in = 0; in < 2; ++in) {
#pragma unroll
      for (int r = 0; r < 8; ++r) {
        out[(size_t)(m0 + im * 16 + r + hi * 8) * EE + n0 + in * 16 + col] =
            accs[im][in][r];
      }
    }
  }
}

// ---------------- launcher ----------------
extern "C" void kernel_launch(void* const* d_in, const int* in_sizes, int n_in,
                              void* d_out, int out_size, void* d_ws, size_t ws_size,
                              hipStream_t stream) {
  const float* x     = (const float*)d_in[0];  // [B,S,E]
  const float* w_qkv = (const float*)d_in[1];  // [E,H,3D] == [1024,3072]
  const float* w_out = (const float*)d_in[2];  // [H,D,E]  == [1024,1024]
  float* out = (float*)d_out;

  const size_t QKV_BYTES = (size_t)BB * HH * SS * DD * sizeof(_Float16);  // 8 MB
  char* ws = (char*)d_ws;
  _Float16* Q  = (_Float16*)(ws);
  _Float16* K  = (_Float16*)(ws + QKV_BYTES);
  _Float16* Vt = (_Float16*)(ws + 2 * QKV_BYTES);
  _Float16* AO = (_Float16*)(ws + 3 * QKV_BYTES);

  qkv_gemm_kernel<<<dim3(HH, MROWS / 32), 192, 0, stream>>>(x, w_qkv, Q, K, Vt);
  flash_attn_kernel<<<dim3(SS / 16, BB * HH), 32, 0, stream>>>(Q, K, Vt, AO);
  out_gemm_kernel<<<dim3(EE / 256, MROWS / 32), 256, 0, stream>>>(AO, w_out, out);
}
